// TfEncoder_69638599737371
// MI455X (gfx1250) — compile-verified, matrix operand order checked by reading, not run on hardware
//
#include <hip/hip_runtime.h>
#include <cstdint>

typedef unsigned short u16;
typedef unsigned int   u32;

typedef __attribute__((ext_vector_type(16))) __bf16 bf16x16;
typedef __attribute__((ext_vector_type(8)))  float  f32x8;

static constexpr int Bn   = 8;
static constexpr int Tn   = 1024;
static constexpr int Cn   = 512;
static constexpr int Hn   = 8;
static constexpr int Dn   = 64;
static constexpr int FFn  = 2048;
static constexpr int NDEPTH = 4;
static constexpr int Mn   = Bn * Tn;   // 8192 rows

struct alignas(16) U4 { u32 x, y, z, w; };

union Frag {
    bf16x16 v;
    U4 q[2];
};

__device__ __forceinline__ u16 f2bf(float f) {
    u32 x = __float_as_uint(f);
    x += 0x7FFFu + ((x >> 16) & 1u);   // round-to-nearest-even
    return (u16)(x >> 16);
}
__device__ __forceinline__ float bf2f(u16 s) {
    return __uint_as_float(((u32)s) << 16);
}

__device__ __forceinline__ f32x8 wmma_bf16(const Frag& a, const Frag& b, f32x8 c) {
    return __builtin_amdgcn_wmma_f32_16x16x32_bf16(
        false, a.v, false, b.v, (short)0, c, false, false);
}

// Async global->LDS 16B copy (per-lane), tracked by ASYNCcnt.
// LDS operand is the raw LDS byte offset = low 32 bits of the generic pointer.
__device__ __forceinline__ void async_ld16(u16* ldsDst, const u16* g) {
    u32 laddr = (u32)(uintptr_t)ldsDst;
    asm volatile("global_load_async_to_lds_b128 %0, %1, off"
                 :: "v"(laddr), "v"(g) : "memory");
}

__device__ __forceinline__ void wait_async0() {
#if __has_builtin(__builtin_amdgcn_s_wait_asynccnt)
    __builtin_amdgcn_s_wait_asynccnt(0);
#else
    asm volatile("s_wait_asynccnt 0" ::: "memory");
#endif
}

// ---------------------------------------------------------------------------
// LayerNorm: fp32 [row, C] -> bf16 [row, C]
// ---------------------------------------------------------------------------
__global__ __launch_bounds__(256) void ln_kernel(
    const float* __restrict__ x, const float* __restrict__ g,
    const float* __restrict__ beta, u16* __restrict__ out)
{
    const int row = blockIdx.x;
    const int tid = threadIdx.x;
    const float* xr = x + (size_t)row * Cn;
    float v0 = xr[tid], v1 = xr[tid + 256];
    float s  = v0 + v1;
    float ss = v0 * v0 + v1 * v1;
    for (int o = 16; o >= 1; o >>= 1) {
        s  += __shfl_xor(s,  o, 32);
        ss += __shfl_xor(ss, o, 32);
    }
    __shared__ float wsum[8], wsq[8];
    const int w = tid >> 5;
    if ((tid & 31) == 0) { wsum[w] = s; wsq[w] = ss; }
    __syncthreads();
    float ts = 0.f, tss = 0.f;
    for (int i = 0; i < 8; i++) { ts += wsum[i]; tss += wsq[i]; }
    const float mu  = ts * (1.f / Cn);
    const float var = tss * (1.f / Cn) - mu * mu;
    const float inv = rsqrtf(var + 1e-5f);
    u16* orow = out + (size_t)row * Cn;
    orow[tid]       = f2bf((v0 - mu) * inv * g[tid]       + beta[tid]);
    orow[tid + 256] = f2bf((v1 - mu) * inv * g[tid + 256] + beta[tid + 256]);
}

// ---------------------------------------------------------------------------
// Weight convert + transpose: fp32 W[K][N] -> bf16 Wt[N][K]
// ---------------------------------------------------------------------------
__global__ __launch_bounds__(256) void cvtT_kernel(
    const float* __restrict__ W, u16* __restrict__ Wt, int K, int N)
{
    const size_t idx = (size_t)blockIdx.x * 256 + threadIdx.x;
    if (idx >= (size_t)K * N) return;
    const int k = (int)(idx / N);
    const int n = (int)(idx % N);
    Wt[(size_t)n * K + k] = f2bf(W[idx]);
}

// ---------------------------------------------------------------------------
// Repack qkv bf16 [B,T,3C] -> Q(scaled) [B,H,T,D], K [B,H,T,D], V^T [B,H,D,T]
// ---------------------------------------------------------------------------
__global__ __launch_bounds__(256) void repack_kernel(
    const u16* __restrict__ qkv, u16* __restrict__ Q,
    u16* __restrict__ Kk, u16* __restrict__ Vt)
{
    const size_t idx = (size_t)blockIdx.x * 256 + threadIdx.x;  // over B*T*C
    if (idx >= (size_t)Mn * Cn) return;
    const int c  = (int)(idx % Cn);
    const size_t bt = idx / Cn;
    const int t = (int)(bt % Tn);
    const int b = (int)(bt / Tn);
    const int h = c >> 6, d = c & 63;
    const size_t src = bt * (3 * Cn);
    const size_t bh  = (size_t)(b * Hn + h);
    const size_t qk_off = (bh * Tn + t) * Dn + d;
    Q[qk_off]  = f2bf(bf2f(qkv[src + c]) * 0.125f);   // 1/sqrt(64)
    Kk[qk_off] = qkv[src + Cn + c];
    Vt[(bh * Dn + d) * Tn + t] = qkv[src + 2 * Cn + c];
}

// ---------------------------------------------------------------------------
// Tiled bf16 WMMA GEMM:  C[M,N] = A[M,K] @ Bt[N,K]^T  (+ bias, epilogues)
// Double-buffered LDS fed by GLOBAL_LOAD_ASYNC_TO_LDS_B128 (ASYNCcnt).
// MODE 0: out bf16 = acc + bias
// MODE 1: out bf16 = gelu(acc + bias)
// MODE 2: outF fp32 += acc + bias   (residual accumulate, in place)
// ---------------------------------------------------------------------------
template <int MODE>
__global__ __launch_bounds__(256) void gemm_bf16_kernel(
    const u16* __restrict__ A, const u16* __restrict__ Bt,
    const float* __restrict__ bias,
    u16* __restrict__ outBf, float* __restrict__ outF,
    int Mdim, int Ndim, int Kdim)
{
    __shared__ u16 lA[2][128 * 48];
    __shared__ u16 lB[2][128 * 48];
    const int tid = threadIdx.x;
    const int w  = tid >> 5, l = tid & 31;
    const int wm = w >> 2, wn = w & 3;        // 2 x 4 wave grid
    const int ln = l & 15, lh = l >> 4;
    const int m0 = blockIdx.y * 128, n0 = blockIdx.x * 128;

    // Per-thread staging slots: 2 chunks of A + 2 chunks of B per k-tile.
    const int row0 = tid >> 2,          c80 = (tid & 3) * 8;
    const int row1 = (tid + 256) >> 2,  c81 = ((tid + 256) & 3) * 8;

    auto issue_tile = [&](int buf, int k0) {
        async_ld16(&lA[buf][row0 * 48 + c80],
                   &A[(size_t)(m0 + row0) * Kdim + k0 + c80]);
        async_ld16(&lA[buf][row1 * 48 + c81],
                   &A[(size_t)(m0 + row1) * Kdim + k0 + c81]);
        async_ld16(&lB[buf][row0 * 48 + c80],
                   &Bt[(size_t)(n0 + row0) * Kdim + k0 + c80]);
        async_ld16(&lB[buf][row1 * 48 + c81],
                   &Bt[(size_t)(n0 + row1) * Kdim + k0 + c81]);
    };

    f32x8 acc[4][2];
    for (int im = 0; im < 4; im++)
        for (int in = 0; in < 2; in++)
            for (int r = 0; r < 8; r++) acc[im][in][r] = 0.f;

    issue_tile(0, 0);
    wait_async0();
    __syncthreads();

    int cur = 0;
    for (int k0 = 0; k0 < Kdim; k0 += 32) {
        if (k0 + 32 < Kdim) issue_tile(cur ^ 1, k0 + 32);   // prefetch next tile

        const u16* cA = lA[cur];
        const u16* cB = lB[cur];
        Frag bfr[2];
        #pragma unroll
        for (int in = 0; in < 2; in++) {
            const int nb = wn * 32 + in * 16 + ln;
            bfr[in].q[0] = *(const U4*)&cB[nb * 48 + lh * 8];
            bfr[in].q[1] = *(const U4*)&cB[nb * 48 + 16 + lh * 8];
        }
        #pragma unroll
        for (int im = 0; im < 4; im++) {
            Frag afr;
            const int mb = wm * 64 + im * 16 + ln;
            afr.q[0] = *(const U4*)&cA[mb * 48 + lh * 8];
            afr.q[1] = *(const U4*)&cA[mb * 48 + 16 + lh * 8];
            #pragma unroll
            for (int in = 0; in < 2; in++)
                acc[im][in] = wmma_bf16(afr, bfr[in], acc[im][in]);
        }

        wait_async0();      // next tile's async copies have landed
        __syncthreads();    // all waves done computing on 'cur', data visible
        cur ^= 1;
    }

    // epilogue: lane l holds column n = base + (l&15), rows r + 8*(l>>4)
    #pragma unroll
    for (int im = 0; im < 4; im++) {
        #pragma unroll
        for (int in = 0; in < 2; in++) {
            const int gn = n0 + wn * 32 + in * 16 + ln;
            const float bv = bias[gn];
            const int mbase = m0 + wm * 64 + im * 16 + lh * 8;
            #pragma unroll
            for (int r = 0; r < 8; r++) {
                float v = acc[im][in][r] + bv;
                const size_t off = (size_t)(mbase + r) * Ndim + gn;
                if (MODE == 0) {
                    outBf[off] = f2bf(v);
                } else if (MODE == 1) {
                    float gelu = 0.5f * v *
                        (1.f + tanhf(0.7978845608f * (v + 0.044715f * v * v * v)));
                    outBf[off] = f2bf(gelu);
                } else {
                    outF[off] += v;
                }
            }
        }
    }
}

// ---------------------------------------------------------------------------
// Flash-style attention. Q pre-scaled. One wave = 16 queries.
// grid = (T/64, H, B), block = 128 (4 waves).
// ---------------------------------------------------------------------------
__global__ __launch_bounds__(128) void attn_kernel(
    const u16* __restrict__ Qs,  // [B*H, T, D]  (scaled by 1/sqrt(D))
    const u16* __restrict__ Kk,  // [B*H, T, D]
    const u16* __restrict__ Vt,  // [B*H, D, T]
    u16* __restrict__ Obf)       // [B, T, C] bf16
{
    __shared__ u16 lP[4][16 * 72];   // per-wave P strip, stride 72 (144B, 16B mult)
    const int w = threadIdx.x >> 5, l = threadIdx.x & 31;
    const int ln = l & 15, lh = l >> 4;
    const int b = blockIdx.z, h = blockIdx.y;
    const size_t bh = (size_t)(b * Hn + h);
    const u16* Qp = Qs + bh * Tn * Dn;
    const u16* Kp = Kk + bh * Tn * Dn;
    const u16* Vp = Vt + bh * Dn * Tn;
    const int qBase = blockIdx.x * 64 + w * 16;

    // Q fragments for both K-steps over D (d 0..31, 32..63); resident all kernel
    Frag qf[2];
    {
        const int m = qBase + ln;
        qf[0].q[0] = *(const U4*)&Qp[m * 64 + lh * 8];
        qf[0].q[1] = *(const U4*)&Qp[m * 64 + 16 + lh * 8];
        qf[1].q[0] = *(const U4*)&Qp[m * 64 + 32 + lh * 8];
        qf[1].q[1] = *(const U4*)&Qp[m * 64 + 48 + lh * 8];
    }

    float rmax[8], rsum[8];
    f32x8 oacc[4];
    for (int r = 0; r < 8; r++) { rmax[r] = -1e30f; rsum[r] = 0.f; }
    for (int jd = 0; jd < 4; jd++)
        for (int r = 0; r < 8; r++) oacc[jd][r] = 0.f;

    u16* myP = lP[w];

    for (int kb = 0; kb < Tn / 64; kb++) {
        // ---- scores for 16 q x 64 keys ----
        f32x8 sacc[4];
        #pragma unroll
        for (int jn = 0; jn < 4; jn++) {
            const int n = kb * 64 + jn * 16 + ln;
            Frag kf0, kf1;
            kf0.q[0] = *(const U4*)&Kp[n * 64 + lh * 8];
            kf0.q[1] = *(const U4*)&Kp[n * 64 + 16 + lh * 8];
            kf1.q[0] = *(const U4*)&Kp[n * 64 + 32 + lh * 8];
            kf1.q[1] = *(const U4*)&Kp[n * 64 + 48 + lh * 8];
            f32x8 z;
            for (int r = 0; r < 8; r++) z[r] = 0.f;
            z = wmma_bf16(qf[0], kf0, z);
            sacc[jn] = wmma_bf16(qf[1], kf1, z);
        }
        // ---- online softmax (rows live in 16-lane halves) ----
        #pragma unroll
        for (int r = 0; r < 8; r++) {
            float mx = sacc[0][r];
            for (int jn = 1; jn < 4; jn++) mx = fmaxf(mx, sacc[jn][r]);
            for (int o = 1; o < 16; o <<= 1) mx = fmaxf(mx, __shfl_xor(mx, o, 32));
            const float nm = fmaxf(rmax[r], mx);
            const float corr = __expf(rmax[r] - nm);
            rmax[r] = nm;
            float ps = 0.f;
            for (int jn = 0; jn < 4; jn++) {
                const float p = __expf(sacc[jn][r] - nm);
                sacc[jn][r] = p;
                ps += p;
            }
            for (int o = 1; o < 16; o <<= 1) ps += __shfl_xor(ps, o, 32);
            rsum[r] = rsum[r] * corr + ps;
            for (int jd = 0; jd < 4; jd++) oacc[jd][r] *= corr;
        }
        // ---- P -> LDS (relayout C-frag -> A-frag) ----
        #pragma unroll
        for (int jn = 0; jn < 4; jn++)
            #pragma unroll
            for (int r = 0; r < 8; r++)
                myP[(r + 8 * lh) * 72 + jn * 16 + ln] = f2bf(sacc[jn][r]);

        Frag pf[2];
        {
            const int m = ln;
            pf[0].q[0] = *(const U4*)&myP[m * 72 + lh * 8];
            pf[0].q[1] = *(const U4*)&myP[m * 72 + 16 + lh * 8];
            pf[1].q[0] = *(const U4*)&myP[m * 72 + 32 + lh * 8];
            pf[1].q[1] = *(const U4*)&myP[m * 72 + 48 + lh * 8];
        }
        // ---- O += P x V ----
        #pragma unroll
        for (int jd = 0; jd < 4; jd++) {
            const int n = jd * 16 + ln;                 // d index
            const u16* vr = &Vp[(size_t)n * Tn + kb * 64];
            Frag vf0, vf1;
            vf0.q[0] = *(const U4*)&vr[lh * 8];
            vf0.q[1] = *(const U4*)&vr[16 + lh * 8];
            vf1.q[0] = *(const U4*)&vr[32 + lh * 8];
            vf1.q[1] = *(const U4*)&vr[48 + lh * 8];
            oacc[jd] = wmma_bf16(pf[0], vf0, oacc[jd]);
            oacc[jd] = wmma_bf16(pf[1], vf1, oacc[jd]);
        }
    }

    // ---- normalize and write O into [B,T,C] bf16 ----
    #pragma unroll
    for (int r = 0; r < 8; r++) {
        const float inv = 1.f / rsum[r];
        const int m = qBase + r + 8 * lh;
        #pragma unroll
        for (int jd = 0; jd < 4; jd++) {
            const int d = jd * 16 + ln;
            Obf[((size_t)(b * Tn + m)) * Cn + h * 64 + d] = f2bf(oacc[jd][r] * inv);
        }
    }
}

// ---------------------------------------------------------------------------
extern "C" void kernel_launch(void* const* d_in, const int* in_sizes, int n_in,
                              void* d_out, int out_size, void* d_ws, size_t ws_size,
                              hipStream_t stream) {
    const float* x    = (const float*)d_in[0];
    const float* Wqkv = (const float*)d_in[1];
    const float* bqkv = (const float*)d_in[2];
    const float* Wo   = (const float*)d_in[3];
    const float* bo   = (const float*)d_in[4];
    const float* W1   = (const float*)d_in[5];
    const float* b1   = (const float*)d_in[6];
    const float* W2   = (const float*)d_in[7];
    const float* b2   = (const float*)d_in[8];
    const float* ln1g = (const float*)d_in[9];
    const float* ln1b = (const float*)d_in[10];
    const float* ln2g = (const float*)d_in[11];
    const float* ln2b = (const float*)d_in[12];

    float* xbuf = (float*)d_out;   // residual stream lives in d_out
    hipMemcpyAsync(xbuf, x, (size_t)Mn * Cn * sizeof(float),
                   hipMemcpyDeviceToDevice, stream);

    char* p = (char*)d_ws;
    u16* act = (u16*)p; p += (size_t)Mn * Cn * 2;          // 8 MiB  (h / attn-out)
    u16* big = (u16*)p; p += (size_t)Mn * FFn * 2;         // 32 MiB (qkv / ffn hidden)
    u16* Qs  = (u16*)p; p += (size_t)Bn * Hn * Tn * Dn * 2; // 8 MiB
    u16* Kk  = (u16*)p; p += (size_t)Bn * Hn * Tn * Dn * 2; // 8 MiB
    u16* Vt  = (u16*)p; p += (size_t)Bn * Hn * Tn * Dn * 2; // 8 MiB
    u16* Wt  = (u16*)p; p += (size_t)FFn * Cn * 2;          // 2 MiB

    for (int i = 0; i < NDEPTH; i++) {
        // --- attention block ---
        ln_kernel<<<Mn, 256, 0, stream>>>(xbuf, ln1g + i * Cn, ln1b + i * Cn, act);
        cvtT_kernel<<<(Cn * 3 * Cn + 255) / 256, 256, 0, stream>>>(
            Wqkv + (size_t)i * Cn * 3 * Cn, Wt, Cn, 3 * Cn);
        gemm_bf16_kernel<0><<<dim3(3 * Cn / 128, Mn / 128), 256, 0, stream>>>(
            act, Wt, bqkv + (size_t)i * 3 * Cn, big, nullptr, Mn, 3 * Cn, Cn);
        repack_kernel<<<(Mn * Cn + 255) / 256, 256, 0, stream>>>(big, Qs, Kk, Vt);
        attn_kernel<<<dim3(Tn / 64, Hn, Bn), 128, 0, stream>>>(Qs, Kk, Vt, act);
        cvtT_kernel<<<(Cn * Cn + 255) / 256, 256, 0, stream>>>(
            Wo + (size_t)i * Cn * Cn, Wt, Cn, Cn);
        gemm_bf16_kernel<2><<<dim3(Cn / 128, Mn / 128), 256, 0, stream>>>(
            act, Wt, bo + (size_t)i * Cn, nullptr, xbuf, Mn, Cn, Cn);

        // --- FFN block ---
        ln_kernel<<<Mn, 256, 0, stream>>>(xbuf, ln2g + i * Cn, ln2b + i * Cn, act);
        cvtT_kernel<<<(Cn * FFn + 255) / 256, 256, 0, stream>>>(
            W1 + (size_t)i * Cn * FFn, Wt, Cn, FFn);
        gemm_bf16_kernel<1><<<dim3(FFn / 128, Mn / 128), 256, 0, stream>>>(
            act, Wt, b1 + (size_t)i * FFn, big, nullptr, Mn, FFn, Cn);
        cvtT_kernel<<<(FFn * Cn + 255) / 256, 256, 0, stream>>>(
            W2 + (size_t)i * FFn * Cn, Wt, FFn, Cn);
        gemm_bf16_kernel<2><<<dim3(Cn / 128, Mn / 128), 256, 0, stream>>>(
            big, Wt, b2 + (size_t)i * Cn, nullptr, xbuf, Mn, Cn, FFn);
    }
}
